// MultiTaskGNN_69337952026660
// MI455X (gfx1250) — matrix operand order repaired and checked by reading, not compiled, not run on hardware
//
#include <hip/hip_runtime.h>
#include <hip/hip_bf16.h>
#include <hip/hip_fp16.h>

typedef __attribute__((ext_vector_type(16))) _Float16 v16h;
typedef __attribute__((ext_vector_type(8)))  _Float16 v8h;
typedef __attribute__((ext_vector_type(8)))  float    v8f;

static constexpr int   NN   = 100000;
static constexpr long  EE   = 1600000;
static constexpr int   INC  = 128;   // in_channels == HEADS*HID
static constexpr int   F1   = 128;   // layer-1 output width (4 heads * 32)
static constexpr int   OUTC = 4;
static constexpr float NEG_SLOPE = 0.2f;
static constexpr int   EPW  = 4;     // edges per wave in gather kernels

// ---- monotone float <-> uint key for atomicMax-based segment max ----
__device__ __forceinline__ unsigned f2k(float f) {
    unsigned b = __float_as_uint(f);
    return (b & 0x80000000u) ? ~b : (b | 0x80000000u);
}
__device__ __forceinline__ float k2f(unsigned k) {
    unsigned b = (k & 0x80000000u) ? (k & 0x7FFFFFFFu) : ~k;
    return __uint_as_float(b);
}

// =====================================================================
// One-shot operand prep: f32 -> f16 row-major copy
// =====================================================================
__global__ __launch_bounds__(256) void f32_to_f16(
    const float* __restrict__ in, _Float16* __restrict__ out, long total)
{
    const long t = (long)blockIdx.x * blockDim.x + threadIdx.x;
    if (t < total) out[t] = (_Float16)in[t];
}

// Pack a 128xC weight pair into WMMA B-fragment order:
//   frag[((ct*4 + ks)*32 + lane)*16 + j] = W[(ks*32 + (lane>=16)*8 + j + (j>=8)*8)*C + ct*16 + (lane&15)]
// Layer 1: C=128, 8 col tiles. Layer 2: C=4 padded to one 16-col tile.
__global__ __launch_bounds__(256) void pack_w_frag128(
    const float* __restrict__ Wl, const float* __restrict__ Wr,
    _Float16* __restrict__ fl, _Float16* __restrict__ fr)
{
    const int t = blockIdx.x * blockDim.x + threadIdx.x;
    if (t >= 8 * 4 * 32) return;
    const int lane = t & 31, ks = (t >> 5) & 3, ct = t >> 7;
    const int kB = (lane >> 4) * 8;
    const int col = ct * 16 + (lane & 15);
    const size_t base = ((size_t)(ct * 4 + ks) * 32 + lane) * 16;
    #pragma unroll
    for (int j = 0; j < 16; ++j) {
        const int k = ks * 32 + kB + j + (j >= 8 ? 8 : 0);
        fl[base + j] = (_Float16)Wl[(size_t)k * F1 + col];
        fr[base + j] = (_Float16)Wr[(size_t)k * F1 + col];
    }
}

__global__ __launch_bounds__(128) void pack_w_frag4(
    const float* __restrict__ Wl, const float* __restrict__ Wr,
    _Float16* __restrict__ fl, _Float16* __restrict__ fr)
{
    const int t = blockIdx.x * blockDim.x + threadIdx.x;
    if (t >= 4 * 32) return;
    const int lane = t & 31, ks = t >> 5;
    const int kB = (lane >> 4) * 8;
    const int col = lane & 15;
    const size_t base = ((size_t)ks * 32 + lane) * 16;
    #pragma unroll
    for (int j = 0; j < 16; ++j) {
        const int k = ks * 32 + kB + j + (j >= 8 ? 8 : 0);
        fl[base + j] = (col < OUTC) ? (_Float16)Wl[(size_t)k * OUTC + col] : (_Float16)0.f;
        fr[base + j] = (col < OUTC) ? (_Float16)Wr[(size_t)k * OUTC + col] : (_Float16)0.f;
    }
}

// Build the WMMA A fragment for one K-step from a row-major f16 row:
// halves j=0..7 at K = base, j=8..15 at K = base+16 (ISA 7.12.2 16-bit A layout).
__device__ __forceinline__ v16h load_a_frag(const _Float16* __restrict__ row, int kbase)
{
    const v8h lo = *(const v8h*)(row + kbase);
    const v8h hi = *(const v8h*)(row + kbase + 16);
    return __builtin_shufflevector(lo, hi, 0, 1, 2, 3, 4, 5, 6, 7,
                                           8, 9, 10, 11, 12, 13, 14, 15);
}

// =====================================================================
// Layer-1 dual GEMM:  Yl = X@Wl + bl,  Yr = X@Wr + br  via packed f16 fragments.
// One wave per 16x16 output tile; block = 8 waves = 8 column tiles of one row tile.
// =====================================================================
__global__ __launch_bounds__(256) void gemm_dual_wmma1(
    const _Float16* __restrict__ Xh,
    const _Float16* __restrict__ FL, const _Float16* __restrict__ FR,
    const float* __restrict__ bl, const float* __restrict__ br,
    float* __restrict__ Yl, float* __restrict__ Yr, int n)
{
    const int ct = threadIdx.x >> 5;        // col tile 0..7
    const int lane = threadIdx.x & 31;
    const int r0 = blockIdx.x * 16;
    if (r0 >= n) return;

    const int row  = r0 + (lane & 15);
    const int kA   = (lane >> 4) * 8;
    const int col  = ct * 16 + (lane & 15);
    const int half = lane >> 4;
    const _Float16* __restrict__ xrow = Xh + (size_t)row * INC;

    v8f accL = {}; v8f accR = {};
    #pragma unroll
    for (int ks = 0; ks < 4; ++ks) {
        const v16h a  = load_a_frag(xrow, ks * 32 + kA);
        const v16h bL = *(const v16h*)(FL + ((size_t)(ct * 4 + ks) * 32 + lane) * 16);
        const v16h bR = *(const v16h*)(FR + ((size_t)(ct * 4 + ks) * 32 + lane) * 16);
        accL = __builtin_amdgcn_wmma_f32_16x16x32_f16(false, a, false, bL, (short)0, accL, false, false);
        accR = __builtin_amdgcn_wmma_f32_16x16x32_f16(false, a, false, bR, (short)0, accR, false, false);
    }
    #pragma unroll
    for (int i = 0; i < 8; ++i) {
        const int m = r0 + i + half * 8;
        if (m < n) {
            Yl[(size_t)m * F1 + col] = accL[i] + bl[col];
            Yr[(size_t)m * F1 + col] = accR[i] + br[col];
        }
    }
}

// =====================================================================
// Layer-2 dual GEMM: Y = H@W + b  (W: 128x4 padded to 16 cols, packed fragments)
// One wave per 16-row tile; block = 8 waves = 8 consecutive row tiles.
// =====================================================================
__global__ __launch_bounds__(256) void gemm_dual_wmma2(
    const _Float16* __restrict__ Hh,
    const _Float16* __restrict__ FL, const _Float16* __restrict__ FR,
    const float* __restrict__ bl, const float* __restrict__ br,
    float* __restrict__ Yl, float* __restrict__ Yr, int n)
{
    const int wave = threadIdx.x >> 5;
    const int lane = threadIdx.x & 31;
    const int r0 = (blockIdx.x * 8 + wave) * 16;
    if (r0 >= n) return;

    const int row  = r0 + (lane & 15);
    const int kA   = (lane >> 4) * 8;
    const int col  = lane & 15;
    const int half = lane >> 4;
    const _Float16* __restrict__ hrow = Hh + (size_t)row * F1;

    v8f accL = {}; v8f accR = {};
    #pragma unroll
    for (int ks = 0; ks < 4; ++ks) {
        const v16h a  = load_a_frag(hrow, ks * 32 + kA);
        const v16h bL = *(const v16h*)(FL + ((size_t)ks * 32 + lane) * 16);
        const v16h bR = *(const v16h*)(FR + ((size_t)ks * 32 + lane) * 16);
        accL = __builtin_amdgcn_wmma_f32_16x16x32_f16(false, a, false, bL, (short)0, accL, false, false);
        accR = __builtin_amdgcn_wmma_f32_16x16x32_f16(false, a, false, bR, (short)0, accR, false, false);
    }
    if (col < OUTC) {
        #pragma unroll
        for (int i = 0; i < 8; ++i) {
            const int m = r0 + i + half * 8;
            if (m < n) {
                Yl[(size_t)m * OUTC + col] = accL[i] + bl[col];
                Yr[(size_t)m * OUTC + col] = accR[i] + br[col];
            }
        }
    }
}

// =====================================================================
// Layer-1 edge scoring: one wave handles EPW consecutive edges, prefetching the
// next edge's gather rows (global_prefetch_b8) while computing the current one.
// lane handles channel `lane` of each of 4 heads (f = h*32+lane).
// =====================================================================
__global__ __launch_bounds__(256) void edge_logits1(
    const int* __restrict__ src, const int* __restrict__ dst,
    const float* __restrict__ Xl, const float* __restrict__ Xr,
    const float* __restrict__ att,           // [4*32]
    float* __restrict__ logits,              // [E*4]
    unsigned* __restrict__ mkey)             // [N*4], zero-init == -inf key
{
    const long gid = (long)blockIdx.x * blockDim.x + threadIdx.x;
    const long e0 = (gid >> 5) * EPW;
    const int lane = (int)(gid & 31);
    if (e0 >= EE) return;

    for (int i = 0; i < EPW; ++i) {
        const long e = e0 + i;
        if (e >= EE) break;
        const int s = src[e], d = dst[e];
        if (e + 1 < EE) {   // prefetch next edge's rows
            const int s2 = src[e + 1], d2 = dst[e + 1];
            __builtin_prefetch(Xl + (size_t)s2 * F1 + lane * 4, 0, 0);
            __builtin_prefetch(Xr + (size_t)d2 * F1 + lane * 4, 0, 0);
        }
        float l[4];
        #pragma unroll
        for (int h = 0; h < 4; ++h) {
            const int f = h * 32 + lane;
            float v = Xl[(size_t)s * F1 + f] + Xr[(size_t)d * F1 + f];
            v = v > 0.f ? v : NEG_SLOPE * v;
            l[h] = v * att[f];
        }
        #pragma unroll
        for (int off = 16; off > 0; off >>= 1) {
            #pragma unroll
            for (int h = 0; h < 4; ++h) l[h] += __shfl_xor(l[h], off, 32);
        }
        if (lane == 0) {
            logits[e * 4 + 0] = l[0]; atomicMax(&mkey[(size_t)d * 4 + 0], f2k(l[0]));
            logits[e * 4 + 1] = l[1]; atomicMax(&mkey[(size_t)d * 4 + 1], f2k(l[1]));
            logits[e * 4 + 2] = l[2]; atomicMax(&mkey[(size_t)d * 4 + 2], f2k(l[2]));
            logits[e * 4 + 3] = l[3]; atomicMax(&mkey[(size_t)d * 4 + 3], f2k(l[3]));
        }
    }
}

// exp(logit - max[dst]) in place, accumulate softmax denominator per (dst, head)
__global__ __launch_bounds__(256) void edge_exp1(
    const int* __restrict__ dst,
    float* __restrict__ exbuf,               // in: logits, out: exp values [E*4]
    const unsigned* __restrict__ mkey,
    float* __restrict__ denom)               // [N*4], zero-init
{
    const long t = (long)blockIdx.x * blockDim.x + threadIdx.x;
    if (t >= EE * 4) return;
    const long e = t >> 2;
    const int h = (int)(t & 3);
    const int d = dst[e];
    const float m = k2f(mkey[(size_t)d * 4 + h]);
    const float ex = __expf(exbuf[t] - m);
    exbuf[t] = ex;
    atomicAdd(&denom[(size_t)d * 4 + h], ex);
}

// scatter: acc[dst] += alpha * xl[src]; wave handles EPW edges with prefetch
__global__ __launch_bounds__(256) void edge_scatter1(
    const int* __restrict__ src, const int* __restrict__ dst,
    const float* __restrict__ Xl,
    const float* __restrict__ exbuf, const float* __restrict__ denom,
    float* __restrict__ acc)                 // [N*128], zero-init
{
    const long gid = (long)blockIdx.x * blockDim.x + threadIdx.x;
    const long e0 = (gid >> 5) * EPW;
    const int lane = (int)(gid & 31);
    if (e0 >= EE) return;

    for (int i = 0; i < EPW; ++i) {
        const long e = e0 + i;
        if (e >= EE) break;
        const int s = src[e], d = dst[e];
        if (e + 1 < EE) {
            const int s2 = src[e + 1];
            __builtin_prefetch(Xl + (size_t)s2 * F1 + lane * 4, 0, 0);
        }
        #pragma unroll
        for (int h = 0; h < 4; ++h) {
            const float alpha = exbuf[e * 4 + h] / denom[(size_t)d * 4 + h];
            const int f = h * 32 + lane;
            atomicAdd(&acc[(size_t)d * F1 + f], alpha * Xl[(size_t)s * F1 + f]);
        }
    }
}

// hh = (f16) elu(acc + bias1)  -- f16 copy feeds the layer-2 WMMA GEMM
__global__ __launch_bounds__(256) void node_bias_elu(
    const float* __restrict__ acc, const float* __restrict__ bias,
    _Float16* __restrict__ hh, long total)
{
    const long t = (long)blockIdx.x * blockDim.x + threadIdx.x;
    if (t >= total) return;
    const float v = acc[t] + bias[t & (F1 - 1)];
    hh[t] = (_Float16)(v > 0.f ? v : (__expf(v) - 1.f));
}

// ---- layer 2 edge phase: 1 head, 4 channels, thread per edge ----
__global__ __launch_bounds__(256) void edge_logits2(
    const int* __restrict__ src, const int* __restrict__ dst,
    const float* __restrict__ Yl, const float* __restrict__ Yr,
    const float* __restrict__ att,           // [4]
    float* __restrict__ logits,              // [E]
    unsigned* __restrict__ mkey)             // [N]
{
    const long e = (long)blockIdx.x * blockDim.x + threadIdx.x;
    if (e >= EE) return;
    const int s = src[e], d = dst[e];
    float lg = 0.f;
    #pragma unroll
    for (int c = 0; c < OUTC; ++c) {
        float v = Yl[(size_t)s * OUTC + c] + Yr[(size_t)d * OUTC + c];
        v = v > 0.f ? v : NEG_SLOPE * v;
        lg += v * att[c];
    }
    logits[e] = lg;
    atomicMax(&mkey[d], f2k(lg));
}

__global__ __launch_bounds__(256) void edge_exp2(
    const int* __restrict__ dst,
    float* __restrict__ exbuf, const unsigned* __restrict__ mkey,
    float* __restrict__ denom)
{
    const long e = (long)blockIdx.x * blockDim.x + threadIdx.x;
    if (e >= EE) return;
    const int d = dst[e];
    const float ex = __expf(exbuf[e] - k2f(mkey[d]));
    exbuf[e] = ex;
    atomicAdd(&denom[d], ex);
}

__global__ __launch_bounds__(256) void edge_scatter2(
    const int* __restrict__ src, const int* __restrict__ dst,
    const float* __restrict__ Yl,
    const float* __restrict__ exbuf, const float* __restrict__ denom,
    float* __restrict__ oacc)                // [N*4]
{
    const long e = (long)blockIdx.x * blockDim.x + threadIdx.x;
    if (e >= EE) return;
    const int s = src[e], d = dst[e];
    const float alpha = exbuf[e] / denom[d];
    #pragma unroll
    for (int c = 0; c < OUTC; ++c)
        atomicAdd(&oacc[(size_t)d * OUTC + c], alpha * Yl[(size_t)s * OUTC + c]);
}

__global__ __launch_bounds__(256) void final_out(
    const float* __restrict__ oacc, const float* __restrict__ bias,
    float* __restrict__ out, long total)
{
    const long t = (long)blockIdx.x * blockDim.x + threadIdx.x;
    if (t >= total) return;
    out[t] = oacc[t] + bias[t & 3];
}

// =====================================================================
extern "C" void kernel_launch(void* const* d_in, const int* in_sizes, int n_in,
                              void* d_out, int out_size, void* d_ws, size_t ws_size,
                              hipStream_t stream) {
    const float* x    = (const float*)d_in[0];
    const int*   eidx = (const int*)  d_in[1];
    const int*   src  = eidx;
    const int*   dst  = eidx + EE;
    const float* W1l  = (const float*)d_in[2];
    const float* b1l  = (const float*)d_in[3];
    const float* W1r  = (const float*)d_in[4];
    const float* b1r  = (const float*)d_in[5];
    const float* att1 = (const float*)d_in[6];
    const float* bias1= (const float*)d_in[7];
    const float* W2l  = (const float*)d_in[8];
    const float* b2l  = (const float*)d_in[9];
    const float* W2r  = (const float*)d_in[10];
    const float* b2r  = (const float*)d_in[11];
    const float* att2 = (const float*)d_in[12];
    const float* bias2= (const float*)d_in[13];
    float* out = (float*)d_out;

    // ---- workspace carve-up (float units; half buffers kept 32B-aligned) ----
    float* ws = (float*)d_ws;
    const size_t NF = (size_t)NN * F1;                 // 12.8M
    size_t o = 0;
    auto align8 = [&]() { o = (o + 7) & ~(size_t)7; }; // 32-byte align
    float*    xl1  = ws + o; o += NF;
    float*    xr1  = ws + o; o += NF;
    float*    acc1 = ws + o; o += NF;
    float*    lg1  = ws + o; o += (size_t)EE * 4;      // logits -> exp (in place)
    unsigned* mk1  = (unsigned*)(ws + o); o += (size_t)NN * 4;
    float*    dn1  = ws + o; o += (size_t)NN * 4;
    float*    yl2  = ws + o; o += (size_t)NN * OUTC;
    float*    yr2  = ws + o; o += (size_t)NN * OUTC;
    float*    lg2  = ws + o; o += (size_t)EE;          // logits -> exp (in place)
    unsigned* mk2  = (unsigned*)(ws + o); o += (size_t)NN;
    float*    dn2  = ws + o; o += (size_t)NN;
    float*    oacc = ws + o; o += (size_t)NN * OUTC;
    align8();
    _Float16* xh   = (_Float16*)(ws + o); o += NF / 2; // N*128 halves
    align8();
    _Float16* hh   = (_Float16*)(ws + o); o += NF / 2;
    align8();
    _Float16* fL1  = (_Float16*)(ws + o); o += (8 * 4 * 32 * 16) / 2;
    align8();
    _Float16* fR1  = (_Float16*)(ws + o); o += (8 * 4 * 32 * 16) / 2;
    align8();
    _Float16* fL2  = (_Float16*)(ws + o); o += (4 * 32 * 16) / 2;
    align8();
    _Float16* fR2  = (_Float16*)(ws + o); o += (4 * 32 * 16) / 2;

    // zero-init atomic accumulators (graph-capture-safe)
    hipMemsetAsync(acc1, 0, NF * sizeof(float), stream);
    hipMemsetAsync(mk1,  0, (size_t)NN * 4 * sizeof(unsigned), stream);
    hipMemsetAsync(dn1,  0, (size_t)NN * 4 * sizeof(float), stream);
    hipMemsetAsync(mk2,  0, (size_t)NN * sizeof(unsigned), stream);
    hipMemsetAsync(dn2,  0, (size_t)NN * sizeof(float), stream);
    hipMemsetAsync(oacc, 0, (size_t)NN * OUTC * sizeof(float), stream);

    const int B = 256;
    const int rowTiles = (NN + 15) / 16;                           // 6250

    // ---- operand prep ----
    f32_to_f16<<<(int)((NF + B - 1) / B), B, 0, stream>>>(x, xh, (long)NF);
    pack_w_frag128<<<4, B, 0, stream>>>(W1l, W1r, fL1, fR1);
    pack_w_frag4<<<1, 128, 0, stream>>>(W2l, W2r, fL2, fR2);

    // ---- layer 1 ----
    gemm_dual_wmma1<<<rowTiles, B, 0, stream>>>(xh, fL1, fR1, b1l, b1r, xl1, xr1, NN);

    const long eWaves = (EE + EPW - 1) / EPW;                      // 400000
    const int  eGatherBlocks = (int)((eWaves * 32 + B - 1) / B);   // 50000
    edge_logits1<<<eGatherBlocks, B, 0, stream>>>(src, dst, xl1, xr1, att1, lg1, mk1);

    edge_exp1<<<(int)((EE * 4 + B - 1) / B), B, 0, stream>>>(dst, lg1, mk1, dn1);

    edge_scatter1<<<eGatherBlocks, B, 0, stream>>>(src, dst, xl1, lg1, dn1, acc1);

    const long nf = (long)NN * F1;
    node_bias_elu<<<(int)((nf + B - 1) / B), B, 0, stream>>>(acc1, bias1, hh, nf);

    // ---- layer 2 ----
    gemm_dual_wmma2<<<(rowTiles + 7) / 8, B, 0, stream>>>(hh, fL2, fR2, b2l, b2r, yl2, yr2, NN);

    const int eBlocks = (int)((EE + B - 1) / B);                   // 6250
    edge_logits2<<<eBlocks, B, 0, stream>>>(src, dst, yl2, yr2, att2, lg2, mk2);
    edge_exp2<<<eBlocks, B, 0, stream>>>(dst, lg2, mk2, dn2);
    edge_scatter2<<<eBlocks, B, 0, stream>>>(src, dst, yl2, lg2, dn2, oacc);

    const long no = (long)NN * OUTC;
    final_out<<<(int)((no + B - 1) / B), B, 0, stream>>>(oacc, bias2, out, no);

    (void)in_sizes; (void)n_in; (void)out_size; (void)ws_size;
}
